// DualAttentionModel_8521215115948
// MI455X (gfx1250) — compile-verified, matrix operand order checked by reading, not run on hardware
//
#include <hip/hip_runtime.h>

// ---------------------------------------------------------------------------
// Model dims
// ---------------------------------------------------------------------------
#define NN 10000
#define EE 160000
#define DN 256
#define DE 128
#define DL 128
#define NH 4
#define STOT (NN + EE)

typedef __attribute__((ext_vector_type(16))) __bf16 v16bf;
typedef __attribute__((ext_vector_type(8)))  float  v8f;

struct Pair16 { uint4 a, b; };

// float -> bf16 (round-to-nearest-even)
__device__ __forceinline__ __bf16 f2bf(float f) {
  unsigned u = __float_as_uint(f);
  unsigned r = u + 0x7fffu + ((u >> 16) & 1u);
  unsigned short h = (unsigned short)(r >> 16);
  __bf16 out;
  __builtin_memcpy(&out, &h, 2);
  return out;
}

// monotonic float<->uint encoding for atomicMax-based segment max
__device__ __forceinline__ unsigned f2ord(float f) {
  unsigned u = __float_as_uint(f);
  return (u & 0x80000000u) ? ~u : (u | 0x80000000u);
}
__device__ __forceinline__ float ord2f(unsigned u) {
  return (u & 0x80000000u) ? __uint_as_float(u & 0x7fffffffu)
                           : __uint_as_float(~u);
}

__device__ __forceinline__ float wave_sum32(float v) {
  for (int m = 1; m < 32; m <<= 1) v += __shfl_xor(v, m, 32);
  return v;
}
__device__ __forceinline__ float group8_sum(float v) {
  v += __shfl_xor(v, 1, 32);
  v += __shfl_xor(v, 2, 32);
  v += __shfl_xor(v, 4, 32);
  return v;
}

// CDNA5 async global->LDS copy (tracked by ASYNCcnt).
// Low 32 bits of a generic shared pointer are the wave-relative LDS address.
__device__ __forceinline__ void async_g2l_b128(void* lds_ptr, const void* gaddr) {
  unsigned ldsoff = (unsigned)(size_t)lds_ptr;
  unsigned long long ga = (unsigned long long)(size_t)gaddr;
  asm volatile("global_load_async_to_lds_b128 %0, %1, off"
               :: "v"(ldsoff), "v"(ga)
               : "memory");
}
__device__ __forceinline__ void wait_async0() {
  asm volatile("s_wait_asynccnt 0" ::: "memory");
}

// ---------------------------------------------------------------------------
// Utility kernels
// ---------------------------------------------------------------------------
__global__ void k_f32_to_bf16(const float* __restrict__ in, __bf16* __restrict__ out,
                              long long n) {
  long long st = (long long)gridDim.x * blockDim.x;
  for (long long i = (long long)blockIdx.x * blockDim.x + threadIdx.x; i < n; i += st)
    out[i] = f2bf(in[i]);
}
__global__ void k_fill_f32(float* __restrict__ p, float v, long long n) {
  long long st = (long long)gridDim.x * blockDim.x;
  for (long long i = (long long)blockIdx.x * blockDim.x + threadIdx.x; i < n; i += st)
    p[i] = v;
}
__global__ void k_fill_u32(unsigned* __restrict__ p, unsigned v, long long n) {
  long long st = (long long)gridDim.x * blockDim.x;
  for (long long i = (long long)blockIdx.x * blockDim.x + threadIdx.x; i < n; i += st)
    p[i] = v;
}

// ---------------------------------------------------------------------------
// Tiled bf16 WMMA GEMM: C[M,Nc] = A[M,K] * W[K,Nc] + bias
// BM=BN=128, BK=32; 256 threads = 8 waves; each wave: 32x64 subtile
// (2x4 tiles of v_wmma_f32_16x16x32_bf16). LDS double-buffered.
// A tile staged via CDNA5 async global->LDS (no transpose needed);
// B tile staged via VGPRs (transposed into LDS).
// Requires: K % 32 == 0, Nc % 128 == 0 (true for all GEMMs in this model).
// ---------------------------------------------------------------------------
#define BM 128
#define BN 128
#define BK 32

__global__ __launch_bounds__(256) void k_gemm_bf16(
    const __bf16* __restrict__ A, const __bf16* __restrict__ W,
    const float* __restrict__ bias, float* __restrict__ C,
    int M, int K, int Nc) {
  __shared__ __align__(16) __bf16 lA[2][BM][BK];
  __shared__ __align__(16) __bf16 lB[2][BN][BK];  // stored transposed: [n][k]

  const int tid  = threadIdx.x;
  const int lane = tid & 31;
  const int wid  = tid >> 5;   // 0..7
  const int wr   = wid & 3;    // 4 row groups of 32
  const int wc   = wid >> 2;   // 2 col groups of 64
  const int rowBase = blockIdx.y * BM;
  const int colBase = blockIdx.x * BN;

  const int la_r = tid >> 1;        // 0..127
  const int la_h = tid & 1;         // which 16-col half of the 32-wide A tile
  const int lb_k = tid & 31;        // k within tile
  const int lb_c = (tid >> 5) * 16; // 16-col group of B

  v8f acc[2][4];
#pragma unroll
  for (int i = 0; i < 2; i++)
#pragma unroll
    for (int j = 0; j < 4; j++)
#pragma unroll
      for (int q = 0; q < 8; q++) acc[i][j][q] = 0.0f;

  const int nkt = K / BK;

  // A rows past M are clamped (stay in-bounds); their garbage only feeds
  // C rows that are never stored.
  const int gr = rowBase + la_r;
  const int grc = (gr < M) ? gr : (M - 1);

  auto loadTile = [&](int kt, int buf) {
    const int k0 = kt * BK;
    // A tile: 128 rows x 32 cols, two async 16B copies per thread
    {
      const __bf16* gp = A + (size_t)grc * K + k0 + la_h * 16;
      __bf16* sp = &lA[buf][la_r][la_h * 16];
      async_g2l_b128(sp, gp);
      async_g2l_b128(sp + 8, gp + 8);
    }
    // B tile: 32 k-rows x 128 cols, stored transposed into lB[c][k]
    {
      const __bf16* gp = W + (size_t)(k0 + lb_k) * Nc + colBase + lb_c;
      uint4 q0 = ((const uint4*)gp)[0];
      uint4 q1 = ((const uint4*)gp)[1];
      __bf16 tmp[16];
      __builtin_memcpy(tmp, &q0, 16);
      __builtin_memcpy(tmp + 8, &q1, 16);
#pragma unroll
      for (int j = 0; j < 16; j++) lB[buf][lb_c + j][lb_k] = tmp[j];
    }
  };

  const int hi  = lane >> 4;  // 0/1: which K-half group this lane holds
  const int l16 = lane & 15;

  loadTile(0, 0);
  wait_async0();
  __syncthreads();

  for (int kt = 0; kt < nkt; ++kt) {
    const int buf = kt & 1;
    if (kt + 1 < nkt) loadTile(kt + 1, buf ^ 1);

    v16bf afr[2], bfr[4];
#pragma unroll
    for (int mt = 0; mt < 2; ++mt) {
      int r = wr * 32 + mt * 16 + l16;
      // A 16-bit fragment layout: lanes 0-15 K={0..7,16..23}, lanes 16-31 K={8..15,24..31}
      uint4 lo  = *(const uint4*)&lA[buf][r][hi * 8];
      uint4 hi4 = *(const uint4*)&lA[buf][r][16 + hi * 8];
      Pair16 p{lo, hi4};
      afr[mt] = __builtin_bit_cast(v16bf, p);
    }
#pragma unroll
    for (int nt = 0; nt < 4; ++nt) {
      int c = wc * 64 + nt * 16 + l16;
      uint4 lo  = *(const uint4*)&lB[buf][c][hi * 8];
      uint4 hi4 = *(const uint4*)&lB[buf][c][16 + hi * 8];
      Pair16 p{lo, hi4};
      bfr[nt] = __builtin_bit_cast(v16bf, p);
    }
#pragma unroll
    for (int mt = 0; mt < 2; ++mt)
#pragma unroll
      for (int nt = 0; nt < 4; ++nt)
        acc[mt][nt] = __builtin_amdgcn_wmma_f32_16x16x32_bf16(
            false, afr[mt], false, bfr[nt], (short)0, acc[mt][nt], false, false);

    // wait for next tile's async copies (issued above, overlapped with WMMAs)
    wait_async0();
    __syncthreads();
  }

  // epilogue: C/D layout: VGPR i -> M = (lane/16)*8 + i, N = lane%16
#pragma unroll
  for (int mt = 0; mt < 2; ++mt) {
#pragma unroll
    for (int nt = 0; nt < 4; ++nt) {
      int col = colBase + wc * 64 + nt * 16 + l16;
      float bv = bias ? bias[col] : 0.0f;
#pragma unroll
      for (int i = 0; i < 8; i++) {
        int row = rowBase + wr * 32 + mt * 16 + hi * 8 + i;
        if (row < M) C[(size_t)row * Nc + col] = acc[mt][nt][i] + bv;
      }
    }
  }
}

// ---------------------------------------------------------------------------
// Fused (gated) residual + LayerNorm: one wave per row.
// gmode: 0 = plain a+delta; 1 = per-head sigmoid(gate[h]); 2 = scalar sigmoid(gate[0])
// ---------------------------------------------------------------------------
__global__ void k_ln_fused(const float* __restrict__ a, const float* __restrict__ delta,
                           const float* __restrict__ gate, int gmode, int Hh,
                           const float* __restrict__ lng, const float* __restrict__ lnb,
                           float* __restrict__ out, __bf16* __restrict__ out_bf,
                           int D, int rows) {
  int row  = (int)(((long long)blockIdx.x * blockDim.x + threadIdx.x) >> 5);
  int lane = threadIdx.x & 31;
  if (row >= rows) return;
  int per = D >> 5;  // 4 (D=128) or 8 (D=256)
  int hw  = D / Hh;
  float gs = (gmode == 2) ? 1.f / (1.f + __expf(-gate[0])) : 1.f;

  float x[8];
  float s = 0.f;
  for (int j = 0; j < per; j++) {
    int d = j * 32 + lane;
    float gv = (gmode == 1) ? 1.f / (1.f + __expf(-gate[d / hw])) : gs;
    float v  = a[(size_t)row * D + d] + gv * delta[(size_t)row * D + d];
    x[j] = v;
    s += v;
  }
  s = wave_sum32(s);
  float mean = s / (float)D;
  float vs = 0.f;
  for (int j = 0; j < per; j++) {
    float t = x[j] - mean;
    vs += t * t;
  }
  vs = wave_sum32(vs);
  float inv = rsqrtf(vs / (float)D + 1e-5f);
  for (int j = 0; j < per; j++) {
    int d = j * 32 + lane;
    float y = (x[j] - mean) * inv * lng[d] + lnb[d];
    out[(size_t)row * D + d] = y;
    if (out_bf) out_bf[(size_t)row * D + d] = f2bf(y);
  }
}

// ---------------------------------------------------------------------------
// Node-attends-edges segment softmax, 3 atomic passes (D=256, H=4, d_nh=64)
// entry j in [0,2E): e = j mod E, node = (j<E ? src : tgt)
// ---------------------------------------------------------------------------
__global__ void k_nattn_scores(const float* __restrict__ nq, const float* __restrict__ ekp,
                               const int* __restrict__ src, const int* __restrict__ tgt,
                               float* __restrict__ scores, unsigned* __restrict__ mxU, int E) {
  long long w = ((long long)blockIdx.x * blockDim.x + threadIdx.x) >> 5;
  int lane = threadIdx.x & 31;
  if (w >= 2LL * E) return;
  int e    = (w < E) ? (int)w : (int)(w - E);
  int node = (w < E) ? src[e] : tgt[e];
  float p = 0.f;
#pragma unroll
  for (int j = 0; j < 8; j++) {
    int d = lane * 8 + j;  // head = lane/8 (d_nh = 64)
    p += nq[(size_t)node * DN + d] * ekp[(size_t)e * DN + d];
  }
  p = group8_sum(p) * 0.125f;  // 1/sqrt(64)
  if ((lane & 7) == 0) {
    int h = lane >> 3;
    scores[w * NH + h] = p;
    atomicMax(&mxU[(size_t)node * NH + h], f2ord(p));
  }
}

__global__ void k_nattn_expsum(float* __restrict__ scores, const unsigned* __restrict__ mxU,
                               float* __restrict__ ssum, const int* __restrict__ src,
                               const int* __restrict__ tgt, int E) {
  long long n  = 2LL * E * NH;
  long long st = (long long)gridDim.x * blockDim.x;
  for (long long i = (long long)blockIdx.x * blockDim.x + threadIdx.x; i < n; i += st) {
    long long j = i >> 2;
    int h = (int)(i & 3);
    int node = (j < E) ? src[(int)j] : tgt[(int)(j - E)];
    float m = ord2f(mxU[(size_t)node * NH + h]);
    float ev = __expf(scores[i] - m);
    scores[i] = ev;
    atomicAdd(&ssum[(size_t)node * NH + h], ev);
  }
}

__global__ void k_nattn_agg(const float* __restrict__ scores, const float* __restrict__ ssum,
                            const float* __restrict__ evp, float* __restrict__ nodeout,
                            const int* __restrict__ src, const int* __restrict__ tgt, int E) {
  long long n  = 2LL * E * DN;
  long long st = (long long)gridDim.x * blockDim.x;
  for (long long i = (long long)blockIdx.x * blockDim.x + threadIdx.x; i < n; i += st) {
    long long j = i >> 8;
    int d = (int)(i & 255);
    int h = d >> 6;
    int e    = (j < E) ? (int)j : (int)(j - E);
    int node = (j < E) ? src[e] : tgt[e];
    float attn = scores[j * NH + h] / (ssum[(size_t)node * NH + h] + 1e-10f);
    atomicAdd(&nodeout[(size_t)node * DN + d], evp[(size_t)e * DN + d] * attn);
  }
}

// ---------------------------------------------------------------------------
// Fused edge-attends-endpoints (2-way softmax) + gate + residual + LN (D=128)
// ---------------------------------------------------------------------------
__global__ void k_edge_attn(const float* __restrict__ eqp, const float* __restrict__ nkp,
                            const float* __restrict__ nvp, const float* __restrict__ edgef,
                            const int* __restrict__ src, const int* __restrict__ tgt,
                            const float* __restrict__ egate, const float* __restrict__ lng,
                            const float* __restrict__ lnb, float* __restrict__ ef1, int E) {
  int e    = (int)(((long long)blockIdx.x * blockDim.x + threadIdx.x) >> 5);
  int lane = threadIdx.x & 31;
  if (e >= E) return;
  int s = src[e], t = tgt[e];
  float ps = 0.f, pt = 0.f;
#pragma unroll
  for (int j = 0; j < 4; j++) {
    int d = lane * 4 + j;  // head = lane/8 (d_eh = 32)
    float q = eqp[(size_t)e * DE + d];
    ps += q * nkp[(size_t)s * DE + d];
    pt += q * nkp[(size_t)t * DE + d];
  }
  ps = group8_sum(ps) * 0.17677669529663687f;  // 1/sqrt(32)
  pt = group8_sum(pt) * 0.17677669529663687f;
  float mx = fmaxf(ps, pt);
  float as = __expf(ps - mx), at = __expf(pt - mx);
  float den = as + at + 1e-10f;
  int h = lane >> 3;
  float g = 1.f / (1.f + __expf(-egate[h]));

  float x[4];
  float sum = 0.f;
#pragma unroll
  for (int j = 0; j < 4; j++) {
    int d = lane * 4 + j;
    float o = (as * nvp[(size_t)s * DE + d] + at * nvp[(size_t)t * DE + d]) / den;
    float v = edgef[(size_t)e * DE + d] + g * o;
    x[j] = v;
    sum += v;
  }
  sum = wave_sum32(sum);
  float mean = sum / (float)DE;
  float vs = 0.f;
#pragma unroll
  for (int j = 0; j < 4; j++) {
    float d0 = x[j] - mean;
    vs += d0 * d0;
  }
  vs = wave_sum32(vs);
  float inv = rsqrtf(vs / (float)DE + 1e-5f);
#pragma unroll
  for (int j = 0; j < 4; j++) {
    int d = lane * 4 + j;
    ef1[(size_t)e * DE + d] = (x[j] - mean) * inv * lng[d] + lnb[d];
  }
}

// ---------------------------------------------------------------------------
// Reconciliation bridge helpers
// ---------------------------------------------------------------------------
__global__ void k_build_edge_ctx(const float* __restrict__ ef1, const float* __restrict__ nf1,
                                 const int* __restrict__ src, const int* __restrict__ tgt,
                                 __bf16* __restrict__ out, int E) {
  int e = blockIdx.x;
  if (e >= E) return;
  int s = src[e], t = tgt[e];
  for (int j = threadIdx.x; j < DE + 2 * DN; j += blockDim.x) {
    float v;
    if (j < DE)            v = ef1[(size_t)e * DE + j];
    else if (j < DE + DN)  v = nf1[(size_t)s * DN + (j - DE)];
    else                   v = nf1[(size_t)t * DN + (j - DE - DN)];
    out[(size_t)e * (DE + 2 * DN) + j] = f2bf(v);
  }
}

__global__ void k_scatter_ef2(const float* __restrict__ ef2, float* __restrict__ nsum,
                              const int* __restrict__ src, const int* __restrict__ tgt, int E) {
  long long n  = 2LL * E * DE;
  long long st = (long long)gridDim.x * blockDim.x;
  for (long long i = (long long)blockIdx.x * blockDim.x + threadIdx.x; i < n; i += st) {
    long long j = i >> 7;
    int d = (int)(i & 127);
    int e    = (j < E) ? (int)j : (int)(j - E);
    int node = (j < E) ? src[e] : tgt[e];
    atomicAdd(&nsum[(size_t)node * DE + d], ef2[(size_t)e * DE + d]);
  }
}
__global__ void k_scatter_cnt(float* __restrict__ cnt, const int* __restrict__ src,
                              const int* __restrict__ tgt, int E) {
  long long st = (long long)gridDim.x * blockDim.x;
  for (long long i = (long long)blockIdx.x * blockDim.x + threadIdx.x; i < 2LL * E; i += st) {
    int node = (i < E) ? src[(int)i] : tgt[(int)(i - E)];
    atomicAdd(&cnt[node], 1.0f);
  }
}

__global__ void k_build_node_ctx(const float* __restrict__ nf1, const float* __restrict__ nsum,
                                 const float* __restrict__ cnt, __bf16* __restrict__ out, int Nn) {
  int n = blockIdx.x;
  if (n >= Nn) return;
  float c = cnt[n] + 1e-10f;
  for (int j = threadIdx.x; j < DN + DE; j += blockDim.x) {
    float v = (j < DN) ? nf1[(size_t)n * DN + j] : nsum[(size_t)n * DE + (j - DN)] / c;
    out[(size_t)n * (DN + DE) + j] = f2bf(v);
  }
}

// ---------------------------------------------------------------------------
// Latent bottleneck bipartite attention (D=128, H=4, d_h=32), 4E entries:
// g=0: k=src,q=N+e; g=1: k=tgt,q=N+e; g=2: k=N+e,q=src; g=3: k=N+e,q=tgt
// ---------------------------------------------------------------------------
__global__ void k_lattn_scores(const float* __restrict__ Q, const float* __restrict__ Kp,
                               const int* __restrict__ src, const int* __restrict__ tgt,
                               float* __restrict__ sc, unsigned* __restrict__ mxU, int E, int Nn) {
  long long w = ((long long)blockIdx.x * blockDim.x + threadIdx.x) >> 5;
  int lane = threadIdx.x & 31;
  if (w >= 4LL * E) return;
  int g = (int)(w / E);
  int e = (int)(w - (long long)g * E);
  int qi, ki;
  if (g == 0)      { ki = src[e];  qi = Nn + e; }
  else if (g == 1) { ki = tgt[e];  qi = Nn + e; }
  else if (g == 2) { ki = Nn + e;  qi = src[e]; }
  else             { ki = Nn + e;  qi = tgt[e]; }
  float p = 0.f;
#pragma unroll
  for (int j = 0; j < 4; j++) {
    int d = lane * 4 + j;  // head = lane/8
    p += Q[(size_t)qi * DL + d] * Kp[(size_t)ki * DL + d];
  }
  p = group8_sum(p) * 0.17677669529663687f;
  if ((lane & 7) == 0) {
    int h = lane >> 3;
    sc[w * NH + h] = p;
    atomicMax(&mxU[(size_t)qi * NH + h], f2ord(p));
  }
}

__global__ void k_lattn_expsum(float* __restrict__ sc, const unsigned* __restrict__ mxU,
                               float* __restrict__ ssum, const int* __restrict__ src,
                               const int* __restrict__ tgt, int E, int Nn) {
  long long n  = 4LL * E * NH;
  long long st = (long long)gridDim.x * blockDim.x;
  for (long long i = (long long)blockIdx.x * blockDim.x + threadIdx.x; i < n; i += st) {
    long long w = i >> 2;
    int h = (int)(i & 3);
    int g = (int)(w / E);
    int e = (int)(w - (long long)g * E);
    int qi;
    if (g == 0 || g == 1) qi = Nn + e;
    else if (g == 2)      qi = src[e];
    else                  qi = tgt[e];
    float m = ord2f(mxU[(size_t)qi * NH + h]);
    float ev = __expf(sc[i] - m);
    sc[i] = ev;
    atomicAdd(&ssum[(size_t)qi * NH + h], ev);
  }
}

__global__ void k_lattn_agg(const float* __restrict__ sc, const float* __restrict__ ssum,
                            const float* __restrict__ V, float* __restrict__ attout,
                            const int* __restrict__ src, const int* __restrict__ tgt,
                            int E, int Nn) {
  long long n  = 4LL * E * DL;
  long long st = (long long)gridDim.x * blockDim.x;
  for (long long i = (long long)blockIdx.x * blockDim.x + threadIdx.x; i < n; i += st) {
    long long w = i >> 7;
    int d = (int)(i & 127);
    int h = d >> 5;  // d_h = 32
    int g = (int)(w / E);
    int e = (int)(w - (long long)g * E);
    int qi, ki;
    if (g == 0)      { ki = src[e];  qi = Nn + e; }
    else if (g == 1) { ki = tgt[e];  qi = Nn + e; }
    else if (g == 2) { ki = Nn + e;  qi = src[e]; }
    else             { ki = Nn + e;  qi = tgt[e]; }
    float attn = sc[w * NH + h] / (ssum[(size_t)qi * NH + h] + 1e-10f);
    atomicAdd(&attout[(size_t)qi * DL + d], V[(size_t)ki * DL + d] * attn);
  }
}

// ---------------------------------------------------------------------------
// Host side
// ---------------------------------------------------------------------------
static inline unsigned gs_blocks(long long n) {
  long long b = (n + 255) / 256;
  if (b > 1048576) b = 1048576;
  return (unsigned)b;
}

static inline unsigned ord_encode_host(float f) {
  union { float f; unsigned u; } cu;
  cu.f = f;
  return (cu.u & 0x80000000u) ? ~cu.u : (cu.u | 0x80000000u);
}

extern "C" void kernel_launch(void* const* d_in, const int* in_sizes, int n_in,
                              void* d_out, int out_size, void* d_ws, size_t ws_size,
                              hipStream_t stream) {
  (void)in_sizes; (void)n_in; (void)out_size; (void)ws_size;

  const float* node_f = (const float*)d_in[0];
  const float* edge_f = (const float*)d_in[1];
  const int*   eidx   = (const int*)d_in[2];
  const int*   src    = eidx;
  const int*   tgt    = eidx + EE;

  // params flattened in jax pytree order (sorted dict keys); base index 3
  auto P = [&](int k) -> const float* { return (const float*)d_in[3 + k]; };
  // cag
  const float* cag_egate = P(0);
  const float* cag_ek_b = P(1);  const float* cag_ek_w = P(2);
  const float* cag_en_b = P(3);  const float* cag_en_g = P(4);
  const float* cag_eq_b = P(5);  const float* cag_eq_w = P(6);
  const float* cag_ev_b = P(7);  const float* cag_ev_w = P(8);
  const float* cag_ngate = P(9);
  const float* cag_nk_b = P(10); const float* cag_nk_w = P(11);
  const float* cag_nn_b = P(12); const float* cag_nn_g = P(13);
  const float* cag_nq_b = P(14); const float* cag_nq_w = P(15);
  const float* cag_nv_b = P(16); const float* cag_nv_w = P(17);
  // rb
  const float* rb_efn_b = P(18); const float* rb_efn_w = P(19);
  const float* rb_en_b = P(20);  const float* rb_en_g = P(21);
  const float* rb_nfe_b = P(22); const float* rb_nfe_w = P(23);
  const float* rb_nn_b = P(24);  const float* rb_nn_g = P(25);
  // slb
  const float* slb_ed_b = P(26); const float* slb_ed_w = P(27);
  const float* slb_egate = P(28);
  const float* slb_en_b = P(29); const float* slb_en_g = P(30);
  const float* slb_eu_b = P(31); const float* slb_eu_w = P(32);
  const float* slb_k_b = P(33);  const float* slb_k_w = P(34);
  const float* slb_nd_b = P(35); const float* slb_nd_w = P(36);
  const float* slb_ngate = P(37);
  const float* slb_nn_b = P(38); const float* slb_nn_g = P(39);
  const float* slb_nu_b = P(40); const float* slb_nu_w = P(41);
  const float* slb_o_b = P(42);  const float* slb_o_w = P(43);
  const float* slb_q_b = P(44);  const float* slb_q_w = P(45);
  const float* slb_sn_b = P(46); const float* slb_sn_g = P(47);
  const float* slb_v_b = P(48);  const float* slb_v_w = P(49);

  // ---- workspace carve-out ----
  char* wp = (char*)d_ws;
  auto alloc = [&](size_t bytes) -> void* {
    void* r = (void*)wp;
    wp += (bytes + 255) & ~(size_t)255;
    return r;
  };
  __bf16* node_bf  = (__bf16*)alloc((size_t)NN * DN * 2);
  __bf16* edge_bf  = (__bf16*)alloc((size_t)EE * DE * 2);
  __bf16* ctx_bf   = (__bf16*)alloc((size_t)EE * (DE + 2 * DN) * 2);
  __bf16* nctx_bf  = (__bf16*)alloc((size_t)NN * (DN + DE) * 2);
  __bf16* nf2_bf   = (__bf16*)alloc((size_t)NN * DN * 2);
  __bf16* ef2_bf   = (__bf16*)alloc((size_t)EE * DE * 2);
  __bf16* comb_bf  = (__bf16*)alloc((size_t)STOT * DL * 2);
  __bf16* atto_bf  = (__bf16*)alloc((size_t)STOT * DL * 2);
  __bf16* comb2_bf = (__bf16*)alloc((size_t)STOT * DL * 2);
  __bf16* wbf      = (__bf16*)alloc((size_t)640 * 256 * 2);

  float* nqp     = (float*)alloc((size_t)NN * DN * 4);
  float* ekp     = (float*)alloc((size_t)EE * DN * 4);
  float* evp     = (float*)alloc((size_t)EE * DN * 4);
  float* scores  = (float*)alloc((size_t)2 * EE * NH * 4);
  unsigned* mxU  = (unsigned*)alloc((size_t)NN * NH * 4);
  float* ssum    = (float*)alloc((size_t)NN * NH * 4);
  float* nodeout = (float*)alloc((size_t)NN * DN * 4);
  float* nf1     = (float*)alloc((size_t)NN * DN * 4);
  float* eqp     = (float*)alloc((size_t)EE * DE * 4);
  float* nkp     = (float*)alloc((size_t)NN * DE * 4);
  float* nvp     = (float*)alloc((size_t)NN * DE * 4);
  float* ef1     = (float*)alloc((size_t)EE * DE * 4);
  float* tmpE    = (float*)alloc((size_t)EE * DE * 4);
  float* ef2     = (float*)alloc((size_t)EE * DE * 4);
  float* nsum    = (float*)alloc((size_t)NN * DE * 4);
  float* cnt     = (float*)alloc((size_t)NN * 4);
  float* tmpN    = (float*)alloc((size_t)NN * DN * 4);
  float* nf2     = (float*)alloc((size_t)NN * DN * 4);
  float* comb    = (float*)alloc((size_t)STOT * DL * 4);
  float* Qb      = (float*)alloc((size_t)STOT * DL * 4);
  float* Kb      = (float*)alloc((size_t)STOT * DL * 4);
  float* Vb      = (float*)alloc((size_t)STOT * DL * 4);
  float* sc      = (float*)alloc((size_t)4 * EE * NH * 4);
  unsigned* mxU2 = (unsigned*)alloc((size_t)STOT * NH * 4);
  float* ss2     = (float*)alloc((size_t)STOT * NH * 4);
  float* attout  = (float*)alloc((size_t)STOT * DL * 4);
  float* oout    = (float*)alloc((size_t)STOT * DL * 4);
  float* comb2   = (float*)alloc((size_t)STOT * DL * 4);
  float* nuout   = (float*)alloc((size_t)NN * DN * 4);
  float* euout   = (float*)alloc((size_t)EE * DE * 4);

  float* nf3 = (float*)d_out;                    // [NN, DN]
  float* ef3 = (float*)d_out + (size_t)NN * DN;  // [EE, DE]

  const dim3 B256(256);
  auto cvt = [&](const float* in, __bf16* out, long long n) {
    k_f32_to_bf16<<<gs_blocks(n), B256, 0, stream>>>(in, out, n);
  };
  auto gemm = [&](const __bf16* A, const float* w, int K, int Nc, const float* b,
                  int M, float* C) {
    long long wn = (long long)K * Nc;
    k_f32_to_bf16<<<gs_blocks(wn), B256, 0, stream>>>(w, wbf, wn);
    dim3 g(Nc / BN, (M + BM - 1) / BM);
    k_gemm_bf16<<<g, B256, 0, stream>>>(A, wbf, b, C, M, K, Nc);
  };

  // ---- inputs to bf16 ----
  cvt(node_f, node_bf, (long long)NN * DN);
  cvt(edge_f, edge_bf, (long long)EE * DE);

  // ==== CrossAttentionGates: nodes over incident edges ====
  gemm(node_bf, cag_nq_w, DN, DN, cag_nq_b, NN, nqp);
  gemm(edge_bf, cag_nk_w, DE, DN, cag_nk_b, EE, ekp);
  gemm(edge_bf, cag_nv_w, DE, DN, cag_nv_b, EE, evp);

  {
    unsigned enc = ord_encode_host(-1e9f);
    k_fill_u32<<<gs_blocks((long long)NN * NH), B256, 0, stream>>>(mxU, enc, (long long)NN * NH);
    k_fill_u32<<<gs_blocks((long long)STOT * NH), B256, 0, stream>>>(mxU2, enc, (long long)STOT * NH);
  }
  k_fill_f32<<<gs_blocks((long long)NN * NH), B256, 0, stream>>>(ssum, 0.f, (long long)NN * NH);
  k_fill_f32<<<gs_blocks((long long)NN * DN), B256, 0, stream>>>(nodeout, 0.f, (long long)NN * DN);

  {
    long long waves = 2LL * EE;
    k_nattn_scores<<<(unsigned)((waves * 32 + 255) / 256), B256, 0, stream>>>(
        nqp, ekp, src, tgt, scores, mxU, EE);
  }
  k_nattn_expsum<<<gs_blocks(2LL * EE * NH), B256, 0, stream>>>(scores, mxU, ssum, src, tgt, EE);
  k_nattn_agg<<<gs_blocks(2LL * EE * DN), B256, 0, stream>>>(scores, ssum, evp, nodeout, src, tgt, EE);

  k_ln_fused<<<(unsigned)(((long long)NN * 32 + 255) / 256), B256, 0, stream>>>(
      node_f, nodeout, cag_ngate, 1, NH, cag_nn_g, cag_nn_b, nf1, (__bf16*)nullptr, DN, NN);

  // edges over endpoints
  gemm(edge_bf, cag_eq_w, DE, DE, cag_eq_b, EE, eqp);
  gemm(node_bf, cag_ek_w, DN, DE, cag_ek_b, NN, nkp);
  gemm(node_bf, cag_ev_w, DN, DE, cag_ev_b, NN, nvp);
  k_edge_attn<<<(unsigned)((EE * 32LL + 255) / 256), B256, 0, stream>>>(
      eqp, nkp, nvp, edge_f, src, tgt, cag_egate, cag_en_g, cag_en_b, ef1, EE);

  // ==== ReconciliationBridge ====
  k_build_edge_ctx<<<EE, B256, 0, stream>>>(ef1, nf1, src, tgt, ctx_bf, EE);
  gemm(ctx_bf, rb_efn_w, DE + 2 * DN, DE, rb_efn_b, EE, tmpE);
  k_ln_fused<<<(unsigned)((EE * 32LL + 255) / 256), B256, 0, stream>>>(
      ef1, tmpE, (const float*)nullptr, 0, NH, rb_en_g, rb_en_b, ef2, ef2_bf, DE, EE);

  k_fill_f32<<<gs_blocks((long long)NN * DE), B256, 0, stream>>>(nsum, 0.f, (long long)NN * DE);
  k_fill_f32<<<gs_blocks((long long)NN), B256, 0, stream>>>(cnt, 0.f, (long long)NN);
  k_scatter_ef2<<<gs_blocks(2LL * EE * DE), B256, 0, stream>>>(ef2, nsum, src, tgt, EE);
  k_scatter_cnt<<<gs_blocks(2LL * EE), B256, 0, stream>>>(cnt, src, tgt, EE);
  k_build_node_ctx<<<NN, B256, 0, stream>>>(nf1, nsum, cnt, nctx_bf, NN);
  gemm(nctx_bf, rb_nfe_w, DN + DE, DN, rb_nfe_b, NN, tmpN);
  k_ln_fused<<<(unsigned)(((long long)NN * 32 + 255) / 256), B256, 0, stream>>>(
      nf1, tmpN, (const float*)nullptr, 0, NH, rb_nn_g, rb_nn_b, nf2, nf2_bf, DN, NN);

  // ==== SharedLatentBottleneck ====
  gemm(nf2_bf, slb_nd_w, DN, DL, slb_nd_b, NN, comb);
  gemm(ef2_bf, slb_ed_w, DE, DL, slb_ed_b, EE, comb + (size_t)NN * DL);
  cvt(comb, comb_bf, (long long)STOT * DL);
  gemm(comb_bf, slb_q_w, DL, DL, slb_q_b, STOT, Qb);
  gemm(comb_bf, slb_k_w, DL, DL, slb_k_b, STOT, Kb);
  gemm(comb_bf, slb_v_w, DL, DL, slb_v_b, STOT, Vb);

  k_fill_f32<<<gs_blocks((long long)STOT * NH), B256, 0, stream>>>(ss2, 0.f, (long long)STOT * NH);
  k_fill_f32<<<gs_blocks((long long)STOT * DL), B256, 0, stream>>>(attout, 0.f, (long long)STOT * DL);
  {
    long long waves = 4LL * EE;
    k_lattn_scores<<<(unsigned)((waves * 32 + 255) / 256), B256, 0, stream>>>(
        Qb, Kb, src, tgt, sc, mxU2, EE, NN);
  }
  k_lattn_expsum<<<gs_blocks(4LL * EE * NH), B256, 0, stream>>>(sc, mxU2, ss2, src, tgt, EE, NN);
  k_lattn_agg<<<gs_blocks(4LL * EE * DL), B256, 0, stream>>>(sc, ss2, Vb, attout, src, tgt, EE, NN);

  cvt(attout, atto_bf, (long long)STOT * DL);
  gemm(atto_bf, slb_o_w, DL, DL, slb_o_b, STOT, oout);
  k_ln_fused<<<(unsigned)(((long long)STOT * 32 + 255) / 256), B256, 0, stream>>>(
      comb, oout, (const float*)nullptr, 0, NH, slb_sn_g, slb_sn_b, comb2, comb2_bf, DL, STOT);

  gemm(comb2_bf, slb_nu_w, DL, DN, slb_nu_b, NN, nuout);
  k_ln_fused<<<(unsigned)(((long long)NN * 32 + 255) / 256), B256, 0, stream>>>(
      nf2, nuout, slb_ngate, 2, NH, slb_nn_g, slb_nn_b, nf3, (__bf16*)nullptr, DN, NN);

  gemm(comb2_bf + (size_t)NN * DL, slb_eu_w, DL, DE, slb_eu_b, EE, euout);
  k_ln_fused<<<(unsigned)((EE * 32LL + 255) / 256), B256, 0, stream>>>(
      ef2, euout, slb_egate, 2, NH, slb_en_g, slb_en_b, ef3, (__bf16*)nullptr, DE, EE);
}
